// GroupedQueryAttention_56272661512446
// MI455X (gfx1250) — matrix operand order repaired and checked by reading, not compiled, not run on hardware
//
#include <hip/hip_runtime.h>

#define HIDDEN   1024
#define NUM_HEADS  16
#define GROUP_NUM   8
#define HEAD_DIM   64
#define SEQ      4096
#define KVDIM     512   // GROUP_NUM * HEAD_DIM

typedef __attribute__((ext_vector_type(16))) __bf16 bf16x16;
typedef __attribute__((ext_vector_type(8)))  float  f32x8;

__device__ __forceinline__ unsigned short f2bf(float f) {
  unsigned int u = __float_as_uint(f);
  u += 0x7FFFu + ((u >> 16) & 1u);           // round to nearest even
  return (unsigned short)(u >> 16);
}

union Frag16 { uint4 q[2]; bf16x16 v; };

__device__ __forceinline__ bf16x16 load_frag(const unsigned short* p0,
                                             const unsigned short* p1) {
  Frag16 u;
  u.q[0] = *reinterpret_cast<const uint4*>(p0);
  u.q[1] = *reinterpret_cast<const uint4*>(p1);
  return u.v;
}

__device__ __forceinline__ f32x8 wmma_bf16(bf16x16 a, bf16x16 b, f32x8 c) {
  return __builtin_amdgcn_wmma_f32_16x16x32_bf16(false, a, false, b,
                                                 (short)0, c, false, false);
}

// ---------------- staging kernels ----------------

__global__ void cvt_bf16_kernel(const float* __restrict__ src,
                                unsigned short* __restrict__ dst, int n) {
  int i = blockIdx.x * blockDim.x + threadIdx.x;
  int stride = gridDim.x * blockDim.x;
  for (; i < n; i += stride) dst[i] = f2bf(src[i]);
}

// src: f32 [K][N] (reduction-major, as in the reference weights)
// dst: bf16 [N][K]  (so WMMA B-fragments are contiguous along K)
__global__ void transpose_bf16_kernel(const float* __restrict__ src,
                                      unsigned short* __restrict__ dst,
                                      int K, int N) {
  int i = blockIdx.x * blockDim.x + threadIdx.x;
  int total = K * N;
  int stride = gridDim.x * blockDim.x;
  for (; i < total; i += stride) {
    int k = i % K;
    int n = i / K;
    dst[(size_t)n * K + k] = f2bf(src[(size_t)k * N + n]);
  }
}

// ---------------- WMMA GEMM: C[M,N] = A[M,K] * Bt[N,K]^T + bias ----------------
// One wave computes a 32x64 tile: 2 A-fragments x 4 B-fragments = 8 wmmas/k-step.
// mode 0: out bf16 Q  layout [h][s][64]   (val * scale, scale = 0.125)
// mode 1: out bf16 K  layout [g][s][64]
// mode 2: out bf16 Vt layout [n][SEQ]  (i.e. V transposed: [g*64+d][s])
// mode 3: out f32 row-major [M][N]
__global__ __launch_bounds__(256) void wmma_gemm_kernel(
    const unsigned short* __restrict__ A,
    const unsigned short* __restrict__ Bt,
    const float* __restrict__ bias,
    void* __restrict__ out,
    int M, int N, int K, int mode, float scale) {
  const int lane = threadIdx.x & 31;
  const int wid  = blockIdx.x * 8 + (threadIdx.x >> 5);
  const int nsupers = N >> 6;
  const int m0 = (wid / nsupers) << 5;   // 32 rows per wave
  const int n0 = (wid % nsupers) << 6;   // 64 cols per wave
  const int half = lane >> 4;
  const int lm   = lane & 15;

  f32x8 acc[2][4];
#pragma unroll
  for (int t = 0; t < 4; ++t) {
    float b = bias[n0 + t * 16 + lm];
#pragma unroll
    for (int i = 0; i < 8; ++i) { acc[0][t][i] = b; acc[1][t][i] = b; }
  }

  const unsigned short* arow0 = A + (size_t)(m0 + lm) * K;
  const unsigned short* arow1 = A + (size_t)(m0 + 16 + lm) * K;
  for (int k0 = 0; k0 < K; k0 += 32) {
    bf16x16 af0 = load_frag(arow0 + k0 + half * 8, arow0 + k0 + 16 + half * 8);
    bf16x16 af1 = load_frag(arow1 + k0 + half * 8, arow1 + k0 + 16 + half * 8);
#pragma unroll
    for (int t = 0; t < 4; ++t) {
      const unsigned short* brow =
          Bt + (size_t)(n0 + t * 16 + lm) * K + k0 + half * 8;
      bf16x16 bf = load_frag(brow, brow + 16);
      acc[0][t] = wmma_bf16(af0, bf, acc[0][t]);
      acc[1][t] = wmma_bf16(af1, bf, acc[1][t]);
    }
  }

#pragma unroll
  for (int mt = 0; mt < 2; ++mt)
#pragma unroll
    for (int t = 0; t < 4; ++t)
#pragma unroll
      for (int r = 0; r < 8; ++r) {
        int m = m0 + mt * 16 + r + 8 * half;
        int n = n0 + t * 16 + lm;
        float v = acc[mt][t][r] * scale;
        if (mode == 3) {
          ((float*)out)[(size_t)m * N + n] = v;
        } else if (mode == 2) {
          ((unsigned short*)out)[(size_t)n * SEQ + m] = f2bf(v);
        } else {  // mode 0 / 1: [head][s][d]
          int hh = n >> 6, d = n & 63;
          ((unsigned short*)out)[((size_t)hh * SEQ + m) * 64 + d] = f2bf(v);
        }
      }
}

// ---------------- flash attention (one wave = one head x 16 query rows) ----------------
// kv processed 64 at a time: 4 score tiles, 8 QK wmmas + 8 PV wmmas per step.

__global__ __launch_bounds__(128) void attn_kernel(
    const unsigned short* __restrict__ Q,   // [H][S][64], pre-scaled by 0.125
    const unsigned short* __restrict__ Kc,  // [G][S][64]
    const unsigned short* __restrict__ Vt,  // [G][64][S]
    unsigned short* __restrict__ ctx,       // [S][HIDDEN] bf16
    int S) {
  __shared__ unsigned short pshare[4 * 16 * 64];   // 16x64 P tile per wave
  const int lane = threadIdx.x & 31;
  const int w    = threadIdx.x >> 5;
  const int gwid = blockIdx.x * 4 + w;
  const int qblocks = S >> 4;
  const int h  = gwid / qblocks;
  const int qb = gwid % qblocks;
  const int g  = h >> 1;          // GROUP_SIZE = 2
  const int q0 = qb << 4;
  const int half = lane >> 4;
  const int lm   = lane & 15;
  unsigned short* pl = pshare + w * (16 * 64);

  const unsigned short* qrow = Q + ((size_t)h * S + q0 + lm) * 64;
  bf16x16 qf0 = load_frag(qrow + half * 8,      qrow + 16 + half * 8);
  bf16x16 qf1 = load_frag(qrow + 32 + half * 8, qrow + 48 + half * 8);

  f32x8 cacc[4];
#pragma unroll
  for (int t = 0; t < 4; ++t)
#pragma unroll
    for (int i = 0; i < 8; ++i) cacc[t][i] = 0.f;
  float mrow[8], lrow[8];
#pragma unroll
  for (int r = 0; r < 8; ++r) { mrow[r] = -1e30f; lrow[r] = 0.f; }

  const unsigned short* kb = Kc + (size_t)g * S * 64;
  const unsigned short* vb = Vt + (size_t)g * 64 * S;

  for (int kv0 = 0; kv0 < S; kv0 += 64) {
    f32x8 s[4];
#pragma unroll
    for (int nt = 0; nt < 4; ++nt)
#pragma unroll
      for (int i = 0; i < 8; ++i) s[nt][i] = 0.f;

    // scores: four 16x16 tiles covering kv0..kv0+63
#pragma unroll
    for (int nt = 0; nt < 4; ++nt) {
      const unsigned short* kr = kb + (size_t)(kv0 + nt * 16 + lm) * 64 + half * 8;
      s[nt] = wmma_bf16(qf0, load_frag(kr,      kr + 16), s[nt]);
      s[nt] = wmma_bf16(qf1, load_frag(kr + 32, kr + 48), s[nt]);
      __builtin_prefetch(kr + 64 * 64, 0, 3);   // next kv block -> global_prefetch_b8
    }

    // online softmax; row r of this lane is m = r + 8*half, spread over 16 lanes
    float alpha[8];
#pragma unroll
    for (int r = 0; r < 8; ++r) {
      float mx = fmaxf(fmaxf(s[0][r], s[1][r]), fmaxf(s[2][r], s[3][r]));
#pragma unroll
      for (int off = 8; off >= 1; off >>= 1)
        mx = fmaxf(mx, __shfl_xor(mx, off, 32));
      float mnew = fmaxf(mrow[r], mx);
      alpha[r]  = __expf(mrow[r] - mnew);
      float p0 = __expf(s[0][r] - mnew);
      float p1 = __expf(s[1][r] - mnew);
      float p2 = __expf(s[2][r] - mnew);
      float p3 = __expf(s[3][r] - mnew);
      s[0][r] = p0; s[1][r] = p1; s[2][r] = p2; s[3][r] = p3;
      float rs = (p0 + p1) + (p2 + p3);
#pragma unroll
      for (int off = 8; off >= 1; off >>= 1)
        rs += __shfl_xor(rs, off, 32);
      lrow[r] = lrow[r] * alpha[r] + rs;
      mrow[r] = mnew;
    }
#pragma unroll
    for (int t = 0; t < 4; ++t)
#pragma unroll
      for (int r = 0; r < 8; ++r) cacc[t][r] *= alpha[r];

    // route P (16x64) through LDS: D-layout store -> A-layout load
#pragma unroll
    for (int r = 0; r < 8; ++r) {
      int m = r + 8 * half;
#pragma unroll
      for (int nt = 0; nt < 4; ++nt)
        pl[m * 64 + nt * 16 + lm] = f2bf(s[nt][r]);
    }
    asm volatile("s_wait_dscnt 0" ::: "memory");  // per-wave LDS in-order RAW
    bf16x16 pf0 = load_frag(pl + lm * 64 + half * 8,
                            pl + lm * 64 + 16 + half * 8);
    bf16x16 pf1 = load_frag(pl + lm * 64 + 32 + half * 8,
                            pl + lm * 64 + 48 + half * 8);

#pragma unroll
    for (int t = 0; t < 4; ++t) {
      const unsigned short* vr = vb + (size_t)(t * 16 + lm) * S + kv0 + half * 8;
      cacc[t] = wmma_bf16(pf0, load_frag(vr, vr + 16), cacc[t]);
      cacc[t] = wmma_bf16(pf1, load_frag(vr + 32, vr + 48), cacc[t]);
    }
  }

#pragma unroll
  for (int t = 0; t < 4; ++t)
#pragma unroll
    for (int r = 0; r < 8; ++r) {
      int m = q0 + r + 8 * half;
      int col = h * 64 + t * 16 + lm;
      ctx[(size_t)m * HIDDEN + col] = f2bf(cacc[t][r] / lrow[r]);
    }
}

// ---------------- host side ----------------

extern "C" void kernel_launch(void* const* d_in, const int* in_sizes, int n_in,
                              void* d_out, int out_size, void* d_ws, size_t ws_size,
                              hipStream_t stream) {
  const float* X  = (const float*)d_in[0];
  const float* Wq = (const float*)d_in[1];
  const float* bq = (const float*)d_in[2];
  const float* Wk = (const float*)d_in[3];
  const float* bk = (const float*)d_in[4];
  const float* Wv = (const float*)d_in[5];
  const float* bv = (const float*)d_in[6];
  const float* Wo = (const float*)d_in[7];
  const float* bo = (const float*)d_in[8];
  float* out = (float*)d_out;

  unsigned short* p = (unsigned short*)d_ws;
  unsigned short* Xbf = p; p += (size_t)SEQ * HIDDEN;
  unsigned short* WqT = p; p += (size_t)HIDDEN * HIDDEN;
  unsigned short* WkT = p; p += (size_t)HIDDEN * KVDIM;
  unsigned short* WvT = p; p += (size_t)HIDDEN * KVDIM;
  unsigned short* WoT = p; p += (size_t)HIDDEN * HIDDEN;
  unsigned short* Qbf = p; p += (size_t)NUM_HEADS * SEQ * HEAD_DIM;
  unsigned short* Kbf = p; p += (size_t)GROUP_NUM * SEQ * HEAD_DIM;
  unsigned short* Vtb = p; p += (size_t)GROUP_NUM * HEAD_DIM * SEQ;
  unsigned short* Ctx = p; p += (size_t)SEQ * HIDDEN;

  cvt_bf16_kernel<<<2048, 256, 0, stream>>>(X, Xbf, SEQ * HIDDEN);
  transpose_bf16_kernel<<<1024, 256, 0, stream>>>(Wq, WqT, HIDDEN, HIDDEN);
  transpose_bf16_kernel<<<512,  256, 0, stream>>>(Wk, WkT, HIDDEN, KVDIM);
  transpose_bf16_kernel<<<512,  256, 0, stream>>>(Wv, WvT, HIDDEN, KVDIM);
  transpose_bf16_kernel<<<1024, 256, 0, stream>>>(Wo, WoT, HIDDEN, HIDDEN);

  // waves = (M/32)*(N/64); 8 waves per 256-thread block
  wmma_gemm_kernel<<<256, 256, 0, stream>>>(Xbf, WqT, bq, Qbf, SEQ, HIDDEN, HIDDEN, 0, 0.125f);
  wmma_gemm_kernel<<<128, 256, 0, stream>>>(Xbf, WkT, bk, Kbf, SEQ, KVDIM,  HIDDEN, 1, 1.0f);
  wmma_gemm_kernel<<<128, 256, 0, stream>>>(Xbf, WvT, bv, Vtb, SEQ, KVDIM,  HIDDEN, 2, 1.0f);

  // 16 heads * 256 q-blocks = 4096 waves; 4 waves per 128-thread block
  attn_kernel<<<1024, 128, 0, stream>>>(Qbf, Kbf, Vtb, Ctx, SEQ);

  wmma_gemm_kernel<<<256, 256, 0, stream>>>(Ctx, WoT, bo, out, SEQ, HIDDEN, HIDDEN, 3, 1.0f);
}
